// ValueNetwork_52475910423172
// MI455X (gfx1250) — matrix operand order repaired and checked by reading, not compiled, not run on hardware
//
#include <hip/hip_runtime.h>
#include <hip/hip_bf16.h>
#include <stdint.h>

// Problem constants from the reference
#define NSEATS     96
#define INPUT_DIM  64
#define SELF_DIM   16
#define HUMAN_DIM  48      // 64-16
#define NX         97      // rows of X = 1 + 96
#define NPAD       112     // 7 * 16
#define NPAD2      128     // padded K for A (m-dim)
#define KPAD       64      // padded 48 -> 64
#define NWAVES     8

typedef __attribute__((ext_vector_type(16))) _Float16 v16h;
typedef __attribute__((ext_vector_type(8)))  float    v8f;
typedef __attribute__((ext_vector_type(4)))  uint32_t v4u;

struct alignas(16) SharedMem {
    _Float16 Xh[NPAD2 * KPAD];      // X row-major, 128x64 (16384 B)
    _Float16 XT[HUMAN_DIM * NPAD2]; // X transposed, 48x128 (12288 B)
    _Float16 wahT[HUMAN_DIM * KPAD];// w_a^T, 48x64, cols>=48 zero (6144 B)
    _Float16 XWh[NPAD * KPAD];      // X @ w_a, 112x64, cols>=48 zero (14336 B)
    _Float16 Ah[NPAD * NPAD2];      // softmax(A), 112x128, zero padded (28672 B)
    union {
        float Sf[NPAD * NPAD];      // scores f32 112x112 (50176 B)
        struct {                    // live only after softmax
            _Float16 Yh[NPAD * KPAD];   // A @ X, 112x64          (off 0)
            _Float16 w1T[KPAD * KPAD];  // w1^T 64x64, cols>=48 0  (off 14336)
            _Float16 h1h[NPAD * KPAD];  // relu(Y @ w1), 112x64    (off 22528)
        } p2;
    } u;
    float ss[16];
    float hmid[64];
    float z0[64];
    float feat[64];
};

// ---- WMMA fragment helpers (A/B in contiguous k order, wave32) --------------------

union FragH { v16h h; v4u q[2]; };

// A-matrix 16x32 f16: lane L: g=L/16, m=L%16; element i -> K=k0+g*16+i.
// One contiguous 32B run -> 2x ds_load_b128.
__device__ __forceinline__ v16h load_a(const _Float16* base, int ld, int row0, int k0, int lane) {
    const int g = (lane >> 4) & 1, m = lane & 15;
    const _Float16* rp = base + (row0 + m) * ld + k0 + g * 16;
    FragH f;
    f.q[0] = *(const v4u*)(rp);
    f.q[1] = *(const v4u*)(rp + 8);
    return f.h;
}

// B-matrix 32x16 f16 from a TRANSPOSED source: B[k][n] = src[(col0+n)*ld + k0+k].
// lane L: g=L/16, n=L%16; element i -> K=k0+g*16+i. One contiguous 32B run.
__device__ __forceinline__ v16h load_bt(const _Float16* base, int ld, int k0, int col0, int lane) {
    const int g = (lane >> 4) & 1, n = lane & 15;
    const _Float16* rp = base + (col0 + n) * ld + k0 + g * 16;
    FragH f;
    f.q[0] = *(const v4u*)(rp);
    f.q[1] = *(const v4u*)(rp + 8);
    return f.h;
}

// C/D f32 16x16: lane L: g=L/16, n=L%16; vgpr r -> row = r + 8g, col = n
__device__ __forceinline__ void store_c_f32(float* dst, int ld, int row0, int col0, int lane, v8f c) {
    const int g = (lane >> 4) & 1, n = lane & 15;
#pragma unroll
    for (int r = 0; r < 8; ++r)
        dst[(row0 + r + g * 8) * ld + col0 + n] = c[r];
}

__device__ __forceinline__ void store_c_f16(_Float16* dst, int ld, int row0, int col0, int lane,
                                            v8f c, bool do_relu) {
    const int g = (lane >> 4) & 1, n = lane & 15;
#pragma unroll
    for (int r = 0; r < 8; ++r) {
        float v = c[r];
        if (do_relu) v = fmaxf(v, 0.f);
        dst[(row0 + r + g * 8) * ld + col0 + n] = (_Float16)v;
    }
}

__device__ __forceinline__ v8f wmma_step(v16h a, v16h b, v8f c) {
    return __builtin_amdgcn_wmma_f32_16x16x32_f16(false, a, false, b, (short)0, c, false, false);
}

// ---------------------------------------------------------------------------

__global__ void __launch_bounds__(256)
value_net_kernel(const float* __restrict__ state,
                 const float* __restrict__ t_w1, const float* __restrict__ t_b1,
                 const float* __restrict__ t_w2, const float* __restrict__ t_b2,
                 const float* __restrict__ w_a,  const float* __restrict__ w1,
                 const float* __restrict__ w2,   const float* __restrict__ v_w1,
                 const float* __restrict__ v_b1, const float* __restrict__ v_w2,
                 const float* __restrict__ v_b2, float* __restrict__ out) {
    __shared__ SharedMem sm;
    const int b    = blockIdx.x;
    const int tid  = threadIdx.x;
    const int lane = tid & 31;
    const int wave = tid >> 5;
    const float* st_b = state + (size_t)b * NSEATS * INPUT_DIM;

    // ---- Phase 0: zero padded LDS buffers; stage w_a^T; load self_state ----
    {
        uint32_t* zp;
        zp = (uint32_t*)sm.Xh;
        for (int i = tid; i < NPAD2 * KPAD / 2; i += 256) zp[i] = 0u;
        zp = (uint32_t*)sm.XT;
        for (int i = tid; i < HUMAN_DIM * NPAD2 / 2; i += 256) zp[i] = 0u;
        zp = (uint32_t*)sm.XWh;
        for (int i = tid; i < NPAD * KPAD / 2; i += 256) zp[i] = 0u;
        zp = (uint32_t*)sm.Ah;
        for (int i = tid; i < NPAD * NPAD2 / 2; i += 256) zp[i] = 0u;
        for (int i = tid; i < HUMAN_DIM * KPAD; i += 256) {
            int j = i >> 6, k = i & 63;   // wahT[j][k] = w_a[k][j]
            sm.wahT[i] = (_Float16)(k < HUMAN_DIM ? w_a[k * HUMAN_DIM + j] : 0.f);
        }
        if (tid < SELF_DIM) sm.ss[tid] = st_b[tid];
    }
    __syncthreads();

    // ---- Phase 1: tiny MLP layer 1 (16->50); stage human rows into X & X^T ----
    if (tid < 50) {
        float acc = t_b1[tid];
#pragma unroll
        for (int i = 0; i < SELF_DIM; ++i) acc += sm.ss[i] * t_w1[i * 50 + tid];
        sm.hmid[tid] = fmaxf(acc, 0.f);
    }
    for (int i = tid; i < NSEATS * HUMAN_DIM; i += 256) {
        int r = i / HUMAN_DIM, c = i % HUMAN_DIM;
        _Float16 v = (_Float16)st_b[r * INPUT_DIM + SELF_DIM + c];
        sm.Xh[(r + 1) * KPAD + c] = v;
        sm.XT[c * NPAD2 + (r + 1)] = v;
    }
    __syncthreads();

    // ---- Phase 2: tiny MLP layer 2 (50->48) -> X row 0 (both layouts) ----
    if (tid < HUMAN_DIM) {
        float acc = t_b2[tid];
#pragma unroll
        for (int j = 0; j < 50; ++j) acc += sm.hmid[j] * t_w2[j * HUMAN_DIM + tid];
        _Float16 v = (_Float16)fmaxf(acc, 0.f);
        sm.Xh[tid] = v;
        sm.XT[tid * NPAD2] = v;
    }
    __syncthreads();

    // ---- Phase 3: XW = X @ w_a  (112x64 x 64x48), tiles 7x3, K=64 ----
    for (int t = wave; t < 7 * 3; t += NWAVES) {
        const int mt = t / 3, nt = t % 3;
        v8f c = {};
#pragma unroll
        for (int kk = 0; kk < KPAD; kk += 32) {
            v16h a = load_a(sm.Xh, KPAD, mt * 16, kk, lane);
            v16h bb = load_bt(sm.wahT, KPAD, kk, nt * 16, lane);
            c = wmma_step(a, bb, c);
        }
        store_c_f16(sm.XWh, KPAD, mt * 16, nt * 16, lane, c, false);
    }
    __syncthreads();

    // ---- Phase 4: S = XW @ X^T  (112x64 x 64x112), tiles 7x7, f32 out ----
    // B operand is X^T, whose "transposed source" is row-major Xh itself.
    for (int t = wave; t < 7 * 7; t += NWAVES) {
        const int mt = t / 7, nt = t % 7;
        v8f c = {};
#pragma unroll
        for (int kk = 0; kk < KPAD; kk += 32) {
            v16h a = load_a(sm.XWh, KPAD, mt * 16, kk, lane);
            v16h bb = load_bt(sm.Xh, KPAD, kk, nt * 16, lane);
            c = wmma_step(a, bb, c);
        }
        store_c_f32(sm.u.Sf, NPAD, mt * 16, nt * 16, lane, c);
    }
    __syncthreads();

    // ---- Phase 5: row softmax over valid cols [0,97) -> Ah (f16) ----
    for (int r = wave; r < NX; r += NWAVES) {
        float* row = &sm.u.Sf[r * NPAD];
        float mx = -3.0e38f;
        for (int m = lane; m < NX; m += 32) mx = fmaxf(mx, row[m]);
#pragma unroll
        for (int o = 16; o; o >>= 1) mx = fmaxf(mx, __shfl_xor(mx, o, 32));
        float sum = 0.f;
        for (int m = lane; m < NX; m += 32) { float e = __expf(row[m] - mx); row[m] = e; sum += e; }
#pragma unroll
        for (int o = 16; o; o >>= 1) sum += __shfl_xor(sum, o, 32);
        const float inv = 1.f / sum;
        for (int m = lane; m < NX; m += 32) sm.Ah[r * NPAD2 + m] = (_Float16)(row[m] * inv);
    }
    __syncthreads();

    // ---- Phase 6: stage w1^T, zero Y (Sf region is now dead) ----
    {
        uint32_t* zy = (uint32_t*)sm.u.p2.Yh;
        for (int i = tid; i < NPAD * KPAD / 2; i += 256) zy[i] = 0u;
        for (int i = tid; i < KPAD * KPAD; i += 256) {
            int j = i >> 6, k = i & 63;   // w1T[j][k] = w1[k][j]
            sm.u.p2.w1T[i] = (_Float16)(k < HUMAN_DIM ? w1[(k << 6) + j] : 0.f);
        }
    }
    __syncthreads();

    // ---- Phase 7: Y = A @ X  (112x128 x 128x48), tiles 7x3, K=128 ----
    // B operand is X (m x h); its transposed source is XT (48x128).
    for (int t = wave; t < 7 * 3; t += NWAVES) {
        const int mt = t / 3, nt = t % 3;
        v8f c = {};
#pragma unroll
        for (int kk = 0; kk < NPAD2; kk += 32) {
            v16h a = load_a(sm.Ah, NPAD2, mt * 16, kk, lane);
            v16h bb = load_bt(sm.XT, NPAD2, kk, nt * 16, lane);
            c = wmma_step(a, bb, c);
        }
        store_c_f16(sm.u.p2.Yh, KPAD, mt * 16, nt * 16, lane, c, false);
    }
    __syncthreads();

    // ---- Phase 8: h1 = relu(Y @ w1)  (112x64 x 64x64), tiles 7x4 ----
    for (int t = wave; t < 7 * 4; t += NWAVES) {
        const int mt = t >> 2, nt = t & 3;
        v8f c = {};
#pragma unroll
        for (int kk = 0; kk < KPAD; kk += 32) {
            v16h a = load_a(sm.u.p2.Yh, KPAD, mt * 16, kk, lane);
            v16h bb = load_bt(sm.u.p2.w1T, KPAD, kk, nt * 16, lane);
            c = wmma_step(a, bb, c);
        }
        store_c_f16(sm.u.p2.h1h, KPAD, mt * 16, nt * 16, lane, c, true);
    }
    __syncthreads();

    // ---- Phase 9: row-0 shortcut: z0 = A[0,:] @ h1 (1x97 @ 97x64) ----
    if (tid < 64) {
        float acc = 0.f;
        for (int m = 0; m < NX; ++m)
            acc += (float)sm.Ah[m] * (float)sm.u.p2.h1h[m * KPAD + tid];
        sm.z0[tid] = acc;
    }
    __syncthreads();

    // ---- Phase 10: feat = relu(z0 @ w2) ----
    if (tid < 64) {
        float acc = 0.f;
#pragma unroll 8
        for (int k = 0; k < 64; ++k) acc += sm.z0[k] * w2[k * 64 + tid];
        sm.feat[tid] = fmaxf(acc, 0.f);
    }
    __syncthreads();

    // ---- Phase 11: hv = relu(feat @ v_w1 + v_b1) ----
    if (tid < 64) {
        float acc = v_b1[tid];
#pragma unroll 8
        for (int k = 0; k < 64; ++k) acc += sm.feat[k] * v_w1[k * 64 + tid];
        sm.hmid[tid] = fmaxf(acc, 0.f);
    }
    __syncthreads();

    // ---- Phase 12: value = hv . v_w2 + v_b2 ----
    if (wave == 0) {
        float p = sm.hmid[lane] * v_w2[lane] + sm.hmid[lane + 32] * v_w2[lane + 32];
#pragma unroll
        for (int o = 16; o; o >>= 1) p += __shfl_xor(p, o, 32);
        if (lane == 0) out[b] = p + v_b2[0];
    }
}

extern "C" void kernel_launch(void* const* d_in, const int* in_sizes, int n_in,
                              void* d_out, int out_size, void* d_ws, size_t ws_size,
                              hipStream_t stream) {
    const float* state = (const float*)d_in[0];
    const float* t_w1  = (const float*)d_in[1];
    const float* t_b1  = (const float*)d_in[2];
    const float* t_w2  = (const float*)d_in[3];
    const float* t_b2  = (const float*)d_in[4];
    const float* w_a   = (const float*)d_in[5];
    const float* w1    = (const float*)d_in[6];
    const float* w2    = (const float*)d_in[7];
    const float* v_w1  = (const float*)d_in[8];
    const float* v_b1  = (const float*)d_in[9];
    const float* v_w2  = (const float*)d_in[10];
    const float* v_b2  = (const float*)d_in[11];
    float* out = (float*)d_out;

    const int batch = in_sizes[0] / (NSEATS * INPUT_DIM);
    value_net_kernel<<<batch, 256, 0, stream>>>(state, t_w1, t_b1, t_w2, t_b2, w_a, w1,
                                                w2, v_w1, v_b1, v_w2, v_b2, out);
}
